// Block_7842610283169
// MI455X (gfx1250) — compile-verified, matrix operand order checked by reading, not run on hardware
//
#include <hip/hip_runtime.h>
#include <hip/hip_bf16.h>
#include <math.h>

typedef __bf16 bf16;
typedef __attribute__((ext_vector_type(16))) __bf16 v16bf;
typedef __attribute__((ext_vector_type(8)))  __bf16 v8bf;
typedef __attribute__((ext_vector_type(8)))  float  v8f;

#define WMMA_BF16(a,b,c) __builtin_amdgcn_wmma_f32_16x16x32_bf16(false,(a),false,(b),(short)0,(c),false,false)

#define B_   2
#define S_   1024
#define D_   1024
#define H_   16
#define HD_  64
#define I_   1024
#define ER_  31
#define KACT 3
#define NT_  (B_*S_)          // 2048 tokens
#define HH_  (H_*HD_)         // 1024

// ---------------------------------------------------------------------------
// WMMA fragment loader.  For 16-bit operands the CDNA5 A/B layout per lane is:
//   lanes 0-15 : K = {k0..k0+7, k0+16..k0+23}
//   lanes 16-31: K = {k0+8..k0+15, k0+24..k0+31}
// i.e. two contiguous 8-element (16B) runs.  A rows / B^T rows are selected by
// the caller via `row` (row = m-index for A, n-index for B when B is stored
// transposed [N x K] row-major).
// ---------------------------------------------------------------------------
__device__ __forceinline__ v16bf load_frag(const bf16* __restrict__ base, int ld, int row, int k0) {
  int lane = threadIdx.x & 31;
  int koff = (lane & 16) >> 1;  // 0 or 8
  const bf16* p = base + (size_t)row * ld + k0 + koff;
  v8bf lo = *(const v8bf*)p;
  v8bf hi = *(const v8bf*)(p + 16);
  v16bf f;
#pragma unroll
  for (int i = 0; i < 8; i++) { f[i] = lo[i]; f[i + 8] = hi[i]; }
  return f;
}

__device__ __forceinline__ float gelu_exact(float x) {
  return 0.5f * x * (1.0f + erff(x * 0.70710678118654752440f));
}

// ---------------------------------------------------------------------------
// fp32 [R x C] -> bf16 transposed [C x R]   (batched over blockIdx.z)
// ---------------------------------------------------------------------------
__global__ void transpose_cvt_kernel(const float* __restrict__ src, bf16* __restrict__ dst,
                                     int R, int C) {
  __shared__ float tile[32][33];
  size_t zb = (size_t)blockIdx.z * (size_t)R * C;
  src += zb; dst += zb;
  int c0 = blockIdx.x * 32, r0 = blockIdx.y * 32;
  int tx = threadIdx.x;
#pragma unroll
  for (int i = threadIdx.y; i < 32; i += 8)
    tile[i][tx] = src[(size_t)(r0 + i) * C + c0 + tx];
  __syncthreads();
#pragma unroll
  for (int i = threadIdx.y; i < 32; i += 8)
    dst[(size_t)(c0 + i) * R + r0 + tx] = (bf16)tile[tx][i];
}

// ---------------------------------------------------------------------------
// RMSNorm: one block per token, writes fp32 + bf16 copies
// ---------------------------------------------------------------------------
__global__ void rmsnorm_kernel(const float* __restrict__ x, const float* __restrict__ gamma,
                               float* __restrict__ of, bf16* __restrict__ oh) {
  int t = blockIdx.x;
  const float* xr = x + (size_t)t * D_;
  float ss = 0.f;
  for (int d = threadIdx.x; d < D_; d += blockDim.x) { float v = xr[d]; ss += v * v; }
#pragma unroll
  for (int m = 16; m >= 1; m >>= 1) ss += __shfl_xor(ss, m, 32);
  __shared__ float red[8];
  __shared__ float rinv_s;
  if ((threadIdx.x & 31) == 0) red[threadIdx.x >> 5] = ss;
  __syncthreads();
  if (threadIdx.x < 32) {
    float v = (threadIdx.x < 8) ? red[threadIdx.x] : 0.f;
#pragma unroll
    for (int m = 4; m >= 1; m >>= 1) v += __shfl_xor(v, m, 32);
    if (threadIdx.x == 0) rinv_s = rsqrtf(v / (float)D_ + 1e-5f);
  }
  __syncthreads();
  float rinv = rinv_s;
  for (int d = threadIdx.x; d < D_; d += blockDim.x) {
    float v = xr[d] * rinv * gamma[d];
    of[(size_t)t * D_ + d] = v;
    oh[(size_t)t * D_ + d] = (bf16)v;
  }
}

// ---------------------------------------------------------------------------
// Main WMMA GEMM:  C[M,N] = A[M,K](bf16) * B (stored as B^T [N,K] bf16)
// block = 4 waves; wave w -> rows [m0+32w, +32) x 64 cols: 8 WMMA tiles that
// share the 4 B fragments (12 b128 loads : 8 WMMAs per K-step).
// ---------------------------------------------------------------------------
enum { EPI_F32 = 0, EPI_GELU_BF16 = 1, EPI_ADD1 = 2, EPI_ADD2 = 3 };

template <int EPI>
__global__ __launch_bounds__(128)
void gemm_wmma_kernel(const bf16* __restrict__ A, const bf16* __restrict__ BT,
                      const float* __restrict__ bias,
                      const float* __restrict__ res1, const float* __restrict__ res2,
                      float* __restrict__ outF, bf16* __restrict__ outH,
                      int M, int N, int K) {
  int wave = threadIdx.x >> 5, lane = threadIdx.x & 31;
  int nl = lane & 15, hb = (lane >> 4) << 3;
  int m0 = blockIdx.y * 128 + wave * 32;
  int n0 = blockIdx.x * 64;
  v8f acc[2][4];
#pragma unroll
  for (int mi = 0; mi < 2; mi++)
#pragma unroll
    for (int j = 0; j < 4; j++) acc[mi][j] = (v8f){0.f,0.f,0.f,0.f,0.f,0.f,0.f,0.f};
  for (int k0 = 0; k0 < K; k0 += 32) {
    v16bf a0 = load_frag(A, K, m0 + nl, k0);
    v16bf a1 = load_frag(A, K, m0 + 16 + nl, k0);
#pragma unroll
    for (int j = 0; j < 4; j++) {
      v16bf b = load_frag(BT, K, n0 + j * 16 + nl, k0);
      acc[0][j] = WMMA_BF16(a0, b, acc[0][j]);
      acc[1][j] = WMMA_BF16(a1, b, acc[1][j]);
    }
  }
#pragma unroll
  for (int mi = 0; mi < 2; mi++) {
#pragma unroll
    for (int j = 0; j < 4; j++) {
      int n = n0 + j * 16 + nl;
      float bn = bias[n];
#pragma unroll
      for (int i = 0; i < 8; i++) {
        int m = m0 + mi * 16 + hb + i;
        size_t off = (size_t)m * N + n;
        float v = acc[mi][j][i] + bn;
        if (EPI == EPI_F32)            outF[off] = v;
        else if (EPI == EPI_GELU_BF16) outH[off] = (bf16)gelu_exact(v);
        else if (EPI == EPI_ADD1)      outF[off] = v + res1[off];
        else                           outF[off] = v + res1[off] + res2[off];
      }
    }
  }
}

// ---------------------------------------------------------------------------
// RoPE on q,k: fp32 [tok, h*64+d] -> bf16 [b,h,s,d]   (reference split-halves)
// ---------------------------------------------------------------------------
__global__ void rope_kernel(const float* __restrict__ q, const float* __restrict__ k,
                            const float* __restrict__ theta,
                            bf16* __restrict__ qr, bf16* __restrict__ kr) {
  int idx = blockIdx.x * blockDim.x + threadIdx.x;
  if (idx >= NT_ * H_ * 32) return;
  int d = idx & 31;
  int h = (idx >> 5) & (H_ - 1);
  int t = idx >> 9;
  int s = t & (S_ - 1);
  int b = t >> 10;
  float ang = (float)s * theta[d];
  float cs = cosf(ang), sn = sinf(ang);
  size_t ib = (size_t)t * HH_ + h * HD_ + 2 * d;
  size_t ob = ((size_t)(b * H_ + h) * S_ + s) * HD_ + d;
  float a1 = q[ib], a2 = q[ib + 1];
  qr[ob]      = (bf16)(a1 * cs - a2 * sn);
  qr[ob + 32] = (bf16)(a1 * sn + a2 * cs);
  float b1 = k[ib], b2 = k[ib + 1];
  kr[ob]      = (bf16)(b1 * cs - b2 * sn);
  kr[ob + 32] = (bf16)(b1 * sn + b2 * cs);
}

// v: fp32 [tok, h*64+d] -> bf16 transposed [b,h,d,s] (contiguous over s)
__global__ void vprep_kernel(const float* __restrict__ v, bf16* __restrict__ vt) {
  int idx = blockIdx.x * blockDim.x + threadIdx.x;
  if (idx >= NT_ * D_) return;
  int d = idx & 63;
  int h = (idx >> 6) & (H_ - 1);
  int t = idx >> 10;
  int s = t & (S_ - 1);
  int b = t >> 10;
  vt[((size_t)(b * H_ + h) * HD_ + d) * S_ + s] = (bf16)v[idx];
}

// ---------------------------------------------------------------------------
// Causal flash attention. grid=(S/64, B*H), block=128 (4 waves).
// Wave handles 16 query rows; iterates 32-key chunks; online softmax.
// ---------------------------------------------------------------------------
__global__ __launch_bounds__(128)
void attn_kernel(const bf16* __restrict__ qr, const bf16* __restrict__ kr,
                 const bf16* __restrict__ vt, bf16* __restrict__ ctx) {
  __shared__ __align__(16) bf16 Pl[4][16][32];
  int wave = threadIdx.x >> 5, lane = threadIdx.x & 31;
  int nl = lane & 15, hb = (lane >> 4) << 3;
  int bh = blockIdx.y;
  int b = bh >> 4, h = bh & (H_ - 1);
  int q0 = blockIdx.x * 64 + wave * 16;
  const bf16* Q  = qr + (size_t)bh * S_ * HD_;
  const bf16* Kk = kr + (size_t)bh * S_ * HD_;
  const bf16* Vt = vt + (size_t)bh * HD_ * S_;

  v16bf qa0 = load_frag(Q, HD_, q0 + nl, 0);
  v16bf qa1 = load_frag(Q, HD_, q0 + nl, 32);
  v8f O[4];
#pragma unroll
  for (int j = 0; j < 4; j++) O[j] = (v8f){0.f,0.f,0.f,0.f,0.f,0.f,0.f,0.f};
  float mrow[8], lrow[8];
#pragma unroll
  for (int i = 0; i < 8; i++) { mrow[i] = -INFINITY; lrow[i] = 0.f; }

  int nch = (q0 + 15) / 32 + 1;
  for (int c = 0; c < nch; c++) {
    int k0c = c * 32;
    v8f s0 = (v8f){0.f,0.f,0.f,0.f,0.f,0.f,0.f,0.f};
    v8f s1 = s0;
    {
      v16bf b0 = load_frag(Kk, HD_, k0c + nl, 0);
      v16bf b1 = load_frag(Kk, HD_, k0c + nl, 32);
      s0 = WMMA_BF16(qa0, b0, s0);
      s0 = WMMA_BF16(qa1, b1, s0);
      v16bf b2 = load_frag(Kk, HD_, k0c + 16 + nl, 0);
      v16bf b3 = load_frag(Kk, HD_, k0c + 16 + nl, 32);
      s1 = WMMA_BF16(qa0, b2, s1);
      s1 = WMMA_BF16(qa1, b3, s1);
    }
    const float scale = 0.125f;   // 1/sqrt(64)
    bool need_mask = (k0c + 31 > q0);
#pragma unroll
    for (int i = 0; i < 8; i++) {
      float a = s0[i] * scale, bb = s1[i] * scale;
      if (need_mask) {
        int qrow = q0 + hb + i;
        if (k0c + nl      > qrow) a  = -INFINITY;
        if (k0c + 16 + nl > qrow) bb = -INFINITY;
      }
      // chunk row max (reduce across the 16 lanes owning this row)
      float cmx = fmaxf(a, bb);
#pragma unroll
      for (int m = 8; m >= 1; m >>= 1) cmx = fmaxf(cmx, __shfl_xor(cmx, m, 32));
      float mn = fmaxf(mrow[i], cmx);
      float alpha = expf(mrow[i] - mn);
      mrow[i] = mn;
      float p0 = expf(a - mn), p1 = expf(bb - mn);
      float rs = p0 + p1;
#pragma unroll
      for (int m = 8; m >= 1; m >>= 1) rs += __shfl_xor(rs, m, 32);
      lrow[i] = lrow[i] * alpha + rs;
#pragma unroll
      for (int dt = 0; dt < 4; dt++) O[dt][i] *= alpha;
      Pl[wave][hb + i][nl]      = (bf16)p0;
      Pl[wave][hb + i][16 + nl] = (bf16)p1;
    }
    asm volatile("s_wait_dscnt 0" ::: "memory");
    v16bf pa = load_frag(&Pl[wave][0][0], 32, nl, 0);
#pragma unroll
    for (int dt = 0; dt < 4; dt++) {
      v16bf vb = load_frag(Vt, S_, dt * 16 + nl, k0c);
      O[dt] = WMMA_BF16(pa, vb, O[dt]);
    }
  }
  size_t tokbase = (size_t)b * S_;
#pragma unroll
  for (int i = 0; i < 8; i++) {
    float inv = 1.0f / lrow[i];
    int m = q0 + hb + i;
    size_t row = (tokbase + m) * HH_ + h * HD_;
#pragma unroll
    for (int dt = 0; dt < 4; dt++)
      ctx[row + dt * 16 + nl] = (bf16)(O[dt][i] * inv);
  }
}

// ---------------------------------------------------------------------------
// Router: one wave per token.  softmax over 31 logits, stable top-3,
// scatter (token, gate) into per-expert lists.
// ---------------------------------------------------------------------------
__global__ void router_kernel(const float* __restrict__ an, const float* __restrict__ rw,
                              const float* __restrict__ rb,
                              int* __restrict__ counts, int* __restrict__ lists,
                              float* __restrict__ gatev) {
  int t = blockIdx.x;
  int e = threadIdx.x;
  float logit = -INFINITY;
  if (e < ER_) {
    float s = rb[e];
    const float* ar = an + (size_t)t * D_;
    for (int d = 0; d < D_; d++) s += ar[d] * rw[d * ER_ + e];
    logit = s;
  }
  float mx = logit;
#pragma unroll
  for (int m = 16; m >= 1; m >>= 1) mx = fmaxf(mx, __shfl_xor(mx, m, 32));
  float ex = (e < ER_) ? expf(logit - mx) : 0.f;
  float sum = ex;
#pragma unroll
  for (int m = 16; m >= 1; m >>= 1) sum += __shfl_xor(sum, m, 32);
  float prob = ex / sum;
  float v = (e < ER_) ? prob : -1.f;
  for (int it = 0; it < KACT; it++) {
    float bv = v; int bi = e;
#pragma unroll
    for (int m = 16; m >= 1; m >>= 1) {
      float ov = __shfl_xor(bv, m, 32);
      int   oi = __shfl_xor(bi, m, 32);
      if (ov > bv || (ov == bv && oi < bi)) { bv = ov; bi = oi; }
    }
    if (e == bi) {
      int slot = atomicAdd(&counts[e], 1);
      lists[e * NT_ + slot] = t;
      gatev[e * NT_ + slot] = prob;
      v = -1.f;
    }
  }
}

__global__ void scan_kernel(const int* __restrict__ counts, int* __restrict__ offs) {
  if (threadIdx.x == 0) {
    int a = 0;
    for (int e = 0; e < ER_; e++) { offs[e] = a; a += counts[e]; }
    offs[ER_] = a;
  }
}

// ---------------------------------------------------------------------------
// Expert FFN layer 1: gather rows of an (bf16) per expert, GEMM with
// re_w1^T[e], +bias, GELU -> compact h buffer.   grid=(I/64, 128, 31)
// ---------------------------------------------------------------------------
__global__ __launch_bounds__(128)
void expert_ffn1_kernel(const bf16* __restrict__ anH, const bf16* __restrict__ w1T,
                        const float* __restrict__ b1,
                        const int* __restrict__ counts, const int* __restrict__ offs,
                        const int* __restrict__ lists,
                        bf16* __restrict__ hbuf) {
  int e = blockIdx.z;
  int cnt = counts[e];
  int m0 = blockIdx.y * 16;
  if (m0 >= cnt) return;
  int wave = threadIdx.x >> 5, lane = threadIdx.x & 31;
  int nl = lane & 15, hb = (lane >> 4) << 3;
  int n0 = blockIdx.x * 64 + wave * 16;
  int r = m0 + nl; if (r >= cnt) r = cnt - 1;
  int tok = lists[e * NT_ + r];
  const bf16* Ar = anH + (size_t)tok * D_;
  const bf16* Wt = w1T + (size_t)e * I_ * D_;
  v8f acc = (v8f){0.f,0.f,0.f,0.f,0.f,0.f,0.f,0.f};
  for (int k0 = 0; k0 < D_; k0 += 32) {
    v16bf a = load_frag(Ar, D_, 0, k0);
    v16bf b = load_frag(Wt, D_, n0 + nl, k0);
    acc = WMMA_BF16(a, b, acc);
  }
  int base = offs[e];
  int n = n0 + nl;
  float bn = b1[(size_t)e * I_ + n];
#pragma unroll
  for (int i = 0; i < 8; i++) {
    int m = m0 + hb + i;
    if (m < cnt)
      hbuf[(size_t)(base + m) * I_ + n] = (bf16)gelu_exact(acc[i] + bn);
  }
}

// ---------------------------------------------------------------------------
// Expert FFN layer 2: h @ re_w2^T[e], gate-scale, atomicAdd into output.
// ---------------------------------------------------------------------------
__global__ __launch_bounds__(128)
void expert_ffn2_kernel(const bf16* __restrict__ hbuf, const bf16* __restrict__ w2T,
                        const float* __restrict__ b2,
                        const int* __restrict__ counts, const int* __restrict__ offs,
                        const int* __restrict__ lists, const float* __restrict__ gatev,
                        float* __restrict__ out) {
  int e = blockIdx.z;
  int cnt = counts[e];
  int m0 = blockIdx.y * 16;
  if (m0 >= cnt) return;
  int wave = threadIdx.x >> 5, lane = threadIdx.x & 31;
  int nl = lane & 15, hb = (lane >> 4) << 3;
  int n0 = blockIdx.x * 64 + wave * 16;
  int base = offs[e];
  const bf16* Ar = hbuf + (size_t)(base + m0) * I_;
  const bf16* Wt = w2T + (size_t)e * D_ * I_;
  v8f acc = (v8f){0.f,0.f,0.f,0.f,0.f,0.f,0.f,0.f};
  for (int k0 = 0; k0 < I_; k0 += 32) {
    v16bf a = load_frag(Ar, I_, nl, k0);
    v16bf b = load_frag(Wt, I_, n0 + nl, k0);
    acc = WMMA_BF16(a, b, acc);
  }
  int n = n0 + nl;
  float bn = b2[(size_t)e * D_ + n];
#pragma unroll
  for (int i = 0; i < 8; i++) {
    int m = m0 + hb + i;
    if (m < cnt) {
      int tok = lists[e * NT_ + m];
      float g = gatev[e * NT_ + m];
      atomicAdd(&out[(size_t)tok * D_ + n], g * (acc[i] + bn));
    }
  }
}

// ---------------------------------------------------------------------------
extern "C" void kernel_launch(void* const* d_in, const int* in_sizes, int n_in,
                              void* d_out, int out_size, void* d_ws, size_t ws_size,
                              hipStream_t stream) {
  (void)in_sizes; (void)n_in; (void)out_size; (void)ws_size;
  const float* x    = (const float*)d_in[0];
  const float* th   = (const float*)d_in[2];
  const float* g1   = (const float*)d_in[3];
  const float* g2   = (const float*)d_in[4];
  const float* q_w  = (const float*)d_in[5];
  const float* q_b  = (const float*)d_in[6];
  const float* k_w  = (const float*)d_in[7];
  const float* k_b  = (const float*)d_in[8];
  const float* v_w  = (const float*)d_in[9];
  const float* v_b  = (const float*)d_in[10];
  const float* o_w  = (const float*)d_in[11];
  const float* o_b  = (const float*)d_in[12];
  const float* rw   = (const float*)d_in[13];
  const float* rb   = (const float*)d_in[14];
  const float* s1w  = (const float*)d_in[15];
  const float* s1b  = (const float*)d_in[16];
  const float* s2w  = (const float*)d_in[17];
  const float* s2b  = (const float*)d_in[18];
  const float* r1w  = (const float*)d_in[19];
  const float* r1b  = (const float*)d_in[20];
  const float* r2w  = (const float*)d_in[21];
  const float* r2b  = (const float*)d_in[22];
  float* out = (float*)d_out;

  // ---- workspace carve ----
  char* p = (char*)d_ws;
  auto alloc = [&](size_t bytes) -> void* {
    void* r = (void*)p;
    p += (bytes + 255) & ~(size_t)255;
    return r;
  };
  const size_t MB = (size_t)1024 * 1024;
  bf16* wqT  = (bf16*)alloc(MB * 2);
  bf16* wkT  = (bf16*)alloc(MB * 2);
  bf16* wvT  = (bf16*)alloc(MB * 2);
  bf16* woT  = (bf16*)alloc(MB * 2);
  bf16* ws1T = (bf16*)alloc(MB * 2);
  bf16* ws2T = (bf16*)alloc(MB * 2);
  bf16* wr1T = (bf16*)alloc((size_t)ER_ * MB * 2);
  bf16* wr2T = (bf16*)alloc((size_t)ER_ * MB * 2);
  float* xnF = (float*)alloc((size_t)NT_ * D_ * 4);
  bf16*  xnH = (bf16*)alloc((size_t)NT_ * D_ * 2);
  float* qF  = (float*)alloc((size_t)NT_ * D_ * 4);
  float* kF  = (float*)alloc((size_t)NT_ * D_ * 4);
  float* vF  = (float*)alloc((size_t)NT_ * D_ * 4);
  bf16*  qrH = (bf16*)alloc((size_t)NT_ * D_ * 2);
  bf16*  krH = (bf16*)alloc((size_t)NT_ * D_ * 2);
  bf16*  vtH = (bf16*)alloc((size_t)NT_ * D_ * 2);
  bf16*  ctxH= (bf16*)alloc((size_t)NT_ * D_ * 2);
  float* aoF = (float*)alloc((size_t)NT_ * D_ * 4);
  float* anF = (float*)alloc((size_t)NT_ * D_ * 4);
  bf16*  anH = (bf16*)alloc((size_t)NT_ * D_ * 2);
  bf16*  t1H = (bf16*)alloc((size_t)NT_ * I_ * 2);
  int*   counts = (int*)alloc(ER_ * 4);
  int*   offs   = (int*)alloc((ER_ + 1) * 4);
  int*   lists  = (int*)alloc((size_t)ER_ * NT_ * 4);
  float* gatev  = (float*)alloc((size_t)ER_ * NT_ * 4);
  bf16*  hbuf   = (bf16*)alloc((size_t)(KACT * NT_ + 16) * I_ * 2);

  dim3 tb(32, 8);
  // weight transpose+convert to bf16 W^T
  transpose_cvt_kernel<<<dim3(32, 32, 1), tb, 0, stream>>>(q_w, wqT, D_, HH_);
  transpose_cvt_kernel<<<dim3(32, 32, 1), tb, 0, stream>>>(k_w, wkT, D_, HH_);
  transpose_cvt_kernel<<<dim3(32, 32, 1), tb, 0, stream>>>(v_w, wvT, D_, HH_);
  transpose_cvt_kernel<<<dim3(32, 32, 1), tb, 0, stream>>>(o_w, woT, HH_, D_);
  transpose_cvt_kernel<<<dim3(32, 32, 1), tb, 0, stream>>>(s1w, ws1T, D_, I_);
  transpose_cvt_kernel<<<dim3(32, 32, 1), tb, 0, stream>>>(s2w, ws2T, I_, D_);
  transpose_cvt_kernel<<<dim3(32, 32, ER_), tb, 0, stream>>>(r1w, wr1T, D_, I_);
  transpose_cvt_kernel<<<dim3(32, 32, ER_), tb, 0, stream>>>(r2w, wr2T, I_, D_);

  // xn = rmsnorm(x, gamma1)
  rmsnorm_kernel<<<NT_, 256, 0, stream>>>(x, g1, xnF, xnH);

  // q,k,v projections  (grid: N/64 x M/128)
  dim3 gg(16, 16);
  gemm_wmma_kernel<EPI_F32><<<gg, 128, 0, stream>>>(xnH, wqT, q_b, nullptr, nullptr, qF, nullptr, NT_, HH_, D_);
  gemm_wmma_kernel<EPI_F32><<<gg, 128, 0, stream>>>(xnH, wkT, k_b, nullptr, nullptr, kF, nullptr, NT_, HH_, D_);
  gemm_wmma_kernel<EPI_F32><<<gg, 128, 0, stream>>>(xnH, wvT, v_b, nullptr, nullptr, vF, nullptr, NT_, HH_, D_);

  rope_kernel<<<(NT_ * H_ * 32) / 256, 256, 0, stream>>>(qF, kF, th, qrH, krH);
  vprep_kernel<<<(NT_ * D_) / 256, 256, 0, stream>>>(vF, vtH);

  // causal flash attention
  attn_kernel<<<dim3(S_ / 64, B_ * H_), 128, 0, stream>>>(qrH, krH, vtH, ctxH);

  // attn_out = ctx @ o_w + o_b + xn
  gemm_wmma_kernel<EPI_ADD1><<<gg, 128, 0, stream>>>(ctxH, woT, o_b, xnF, nullptr, aoF, nullptr, NT_, D_, HH_);

  // an = rmsnorm(attn_out, gamma2)
  rmsnorm_kernel<<<NT_, 256, 0, stream>>>(aoF, g2, anF, anH);

  // shared MLP: t1 = gelu(an@sh_w1+b1); out = t1@sh_w2 + b2 + an + attn_out
  gemm_wmma_kernel<EPI_GELU_BF16><<<gg, 128, 0, stream>>>(anH, ws1T, s1b, nullptr, nullptr, nullptr, t1H, NT_, I_, D_);
  gemm_wmma_kernel<EPI_ADD2><<<gg, 128, 0, stream>>>(t1H, ws2T, s2b, anF, aoF, out, nullptr, NT_, D_, I_);

  // router + top-3 gating
  hipMemsetAsync(counts, 0, ER_ * sizeof(int), stream);
  router_kernel<<<NT_, 32, 0, stream>>>(anF, rw, rb, counts, lists, gatev);
  scan_kernel<<<1, 32, 0, stream>>>(counts, offs);

  // routed experts (grouped GEMMs), atomically added onto out
  expert_ffn1_kernel<<<dim3(I_ / 64, NT_ / 16, ER_), 128, 0, stream>>>(anH, wr1T, r1b, counts, offs, lists, hbuf);
  expert_ffn2_kernel<<<dim3(D_ / 64, NT_ / 16, ER_), 128, 0, stream>>>(hbuf, wr2T, r2b, counts, offs, lists, gatev, out);
}
